// Attention_71957882077649
// MI455X (gfx1250) — matrix operand order repaired and checked by reading, not compiled, not run on hardware
//
#include <hip/hip_runtime.h>
#include <hip/hip_bf16.h>
#include <cstdint>
#include <cstddef>

// ---------------------------------------------------------------------------
// XLNet two-stream relative attention for gfx1250 (MI455X), f16 WMMA pipeline.
// ---------------------------------------------------------------------------

typedef _Float16 v16h __attribute__((ext_vector_type(16)));
typedef _Float16 v8h  __attribute__((ext_vector_type(8)));
typedef float    v8f  __attribute__((ext_vector_type(8)));

constexpr int cB = 4, cQ = 512, cM = 512, cP = 128, cHID = 1024, cH = 16, cS = 64;
constexpr int cR = 1024;      // M + Q
constexpr int cPOS = 1536;    // M + 2Q
constexpr float kScale = 0.125f;    // 1/sqrt(64)
constexpr float kNegInf = -1e9f;

__device__ __forceinline__ v8f wmma_f16(v16h a, v16h b, v8f c) {
  return __builtin_amdgcn_wmma_f32_16x16x32_f16(false, a, false, b, (short)0, c,
                                                false, false);
}

__device__ __forceinline__ v16h pack16(v8h lo, v8h hi8) {
  v16h r;
#pragma unroll
  for (int e = 0; e < 8; ++e) { r[e] = lo[e]; r[8 + e] = hi8[e]; }
  return r;
}

// A fragment (f16 16x32): base is [row][k] with `pitch` halves per row.
// ISA layout: lanes 0-15 M=lane K={k0+hi8..}, chunks at k0+hi*8 and k0+16+hi*8.
__device__ __forceinline__ v16h load_A_frag(const _Float16* base, int pitch,
                                            int k0, int lane) {
  int mm = lane & 15, hh = lane >> 4;
  const _Float16* p = base + mm * pitch + k0 + hh * 8;
  v8h lo = *(const v8h*)p;
  v8h hi8 = *(const v8h*)(p + 16);
  return pack16(lo, hi8);
}

__device__ __forceinline__ v16h load_A_frag_bias(const _Float16* base, int pitch,
                                                 int k0, int lane,
                                                 const _Float16* bias) {
  int mm = lane & 15, hh = lane >> 4;
  const _Float16* p = base + mm * pitch + k0 + hh * 8;
  const _Float16* q = bias + k0 + hh * 8;
  v16h r;
#pragma unroll
  for (int e = 0; e < 8; ++e) {
    r[e]     = (_Float16)((float)p[e]      + (float)q[e]);
    r[8 + e] = (_Float16)((float)p[16 + e] + (float)q[16 + e]);
  }
  return r;
}

// B fragment (f16 32x16): baseT is [n][k] (B transposed), `pitch` halves per row.
// ISA layout: lane holds column n=lane&15, chunks at k0+hi*8 and k0+16+hi*8.
__device__ __forceinline__ v16h load_B_frag(const _Float16* baseT, int pitch,
                                            int k0, int lane) {
  int nn = lane & 15, hh = lane >> 4;
  const _Float16* p = baseT + nn * pitch + k0 + hh * 8;
  v8h lo = *(const v8h*)p;
  v8h hi8 = *(const v8h*)(p + 16);
  return pack16(lo, hi8);
}

// ---------------------------------------------------------------------------
// Elementwise helpers
// ---------------------------------------------------------------------------
__global__ void k_cvt(const float* __restrict__ in, _Float16* __restrict__ out,
                      size_t n) {
  size_t i = (size_t)blockIdx.x * blockDim.x + threadIdx.x;
  size_t stride = (size_t)gridDim.x * blockDim.x;
  for (; i < n; i += stride) out[i] = (_Float16)in[i];
}

__global__ void k_concat(const float* __restrict__ mems,
                         const float* __restrict__ cont,
                         _Float16* __restrict__ ctx, size_t n) {
  size_t i = (size_t)blockIdx.x * blockDim.x + threadIdx.x;
  size_t stride = (size_t)gridDim.x * blockDim.x;
  for (; i < n; i += stride) {
    int d = (int)(i % cHID);
    int l = (int)((i / cHID) % cR);
    int b = (int)(i / ((size_t)cHID * cR));
    float v = (l < cM) ? mems[((size_t)b * cM + l) * cHID + d]
                       : cont[((size_t)b * cQ + (l - cM)) * cHID + d];
    ctx[i] = (_Float16)v;
  }
}

__global__ void k_tpos(const float* __restrict__ tm, int* __restrict__ tpos) {
  int t = blockIdx.x * blockDim.x + threadIdx.x;
  if (t >= cB * cP) return;
  const float* row = tm + (size_t)t * cQ;
  int idx = 0;
  for (int q = 0; q < cQ; ++q)
    if (row[q] > 0.5f) idx = q;
  tpos[t] = idx;
}

// ---------------------------------------------------------------------------
// Generic f16 GEMM: C[M,N] = A[M,K] * B[K,N].  M,N multiples of 64, K mult 32.
// Block = 128 threads (4 waves); 64x64 output tile; wave w -> 16 M-rows.
// ---------------------------------------------------------------------------
template <int F32OUT>
__global__ void __launch_bounds__(128) k_gemm(const _Float16* __restrict__ A,
                                              const _Float16* __restrict__ Bm,
                                              void* __restrict__ Cout,
                                              int M, int N, int K) {
  constexpr int GP = 40;  // LDS pitch (halves) for 32-wide k tiles
  __shared__ __align__(16) _Float16 sA[64 * GP];
  __shared__ __align__(16) _Float16 sB[64 * GP];  // transposed [n][k]
  const int tid = threadIdx.x, lane = tid & 31, w = tid >> 5;
  const int m0 = blockIdx.y * 64, n0 = blockIdx.x * 64;
  v8f acc[4] = {};
  for (int k0 = 0; k0 < K; k0 += 32) {
    __syncthreads();
    {  // stage A: 64x32 halves, 16 contiguous per thread
      int idx = tid * 16;
      int r = idx >> 5, c = idx & 31;
      const _Float16* g = A + (size_t)(m0 + r) * K + k0 + c;
      *(v8h*)&sA[r * GP + c]     = *(const v8h*)g;
      *(v8h*)&sA[r * GP + c + 8] = *(const v8h*)(g + 8);
    }
    {  // stage B transposed: thread reads 16 contiguous n of one k row
      int idx = tid * 16;
      int kr = idx >> 6, c = idx & 63;
      const _Float16* g = Bm + (size_t)(k0 + kr) * N + n0 + c;
#pragma unroll
      for (int j = 0; j < 16; ++j) sB[(c + j) * GP + kr] = g[j];
    }
    __syncthreads();
    v16h a = load_A_frag(&sA[(w * 16) * GP], GP, 0, lane);
#pragma unroll
    for (int t = 0; t < 4; ++t) {
      v16h bf = load_B_frag(&sB[(t * 16) * GP], GP, 0, lane);
      acc[t] = wmma_f16(a, bf, acc[t]);
    }
  }
  const int mlc = lane & 15, hl = lane >> 4;
#pragma unroll
  for (int t = 0; t < 4; ++t) {
#pragma unroll
    for (int v = 0; v < 8; ++v) {
      int row = m0 + w * 16 + v + hl * 8;
      int col = n0 + t * 16 + mlc;
      if (F32OUT)
        ((float*)Cout)[(size_t)row * N + col] = acc[t][v];
      else
        ((_Float16*)Cout)[(size_t)row * N + col] = (_Float16)acc[t][v];
    }
  }
}

// ---------------------------------------------------------------------------
// PosQ[b,h,p,rp] = sum_s (QQ[b,p,h,s] + pb[h,s]) * KP[b,rp,h,s]
// grid: (POS/64, P/64, B*H), block 128.
// ---------------------------------------------------------------------------
__global__ void __launch_bounds__(128) k_posq(const _Float16* __restrict__ QQ,
                                              const _Float16* __restrict__ KPp,
                                              const float* __restrict__ pb,
                                              _Float16* __restrict__ PosQ) {
  constexpr int PI = 72;
  __shared__ __align__(16) _Float16 sA[64 * PI];  // [p][s] with bias
  __shared__ __align__(16) _Float16 sB[64 * PI];  // [rp][s]  (== B transposed)
  const int tid = threadIdx.x, lane = tid & 31, w = tid >> 5;
  const int bz = blockIdx.z, b = bz >> 4, h = bz & 15;
  const int p0 = blockIdx.y * 64, rp0 = blockIdx.x * 64;
  {
    int idx = tid * 32;
    int r = idx >> 6, c = idx & 63;
    const _Float16* gq = QQ + ((size_t)(b * cP + p0 + r)) * cHID + h * cS + c;
    const _Float16* gk = KPp + ((size_t)(b * cPOS + rp0 + r)) * cHID + h * cS + c;
    for (int j = 0; j < 32; ++j) {
      sA[r * PI + c + j] = (_Float16)((float)gq[j] + pb[h * cS + c + j]);
      sB[r * PI + c + j] = gk[j];
    }
  }
  __syncthreads();
  v8f acc[4] = {};
#pragma unroll
  for (int kk = 0; kk < 2; ++kk) {
    v16h a = load_A_frag(&sA[(w * 16) * PI], PI, kk * 32, lane);
#pragma unroll
    for (int t = 0; t < 4; ++t) {
      v16h bf = load_B_frag(&sB[(t * 16) * PI], PI, kk * 32, lane);
      acc[t] = wmma_f16(a, bf, acc[t]);
    }
  }
  const int mlc = lane & 15, hl = lane >> 4;
#pragma unroll
  for (int t = 0; t < 4; ++t) {
#pragma unroll
    for (int v = 0; v < 8; ++v) {
      size_t row = ((size_t)(b * cH + h)) * cP + p0 + w * 16 + v + hl * 8;
      PosQ[row * cPOS + rp0 + t * 16 + mlc] = (_Float16)acc[t][v];
    }
  }
}

// ---------------------------------------------------------------------------
// Content-stream attention, flash style, pos logits via in-kernel band GEMM
// + rel-shift gather.  grid: (Q/64, H, B), block 128 (4 waves, 16 q-rows each).
// ---------------------------------------------------------------------------
__global__ void __launch_bounds__(128) k_attn_content(
    const _Float16* __restrict__ QC, const _Float16* __restrict__ KC,
    const _Float16* __restrict__ Vp, const _Float16* __restrict__ KPp,
    const float* __restrict__ cb, const float* __restrict__ pb,
    const float* __restrict__ sb, const float* __restrict__ segenc,
    const unsigned char* __restrict__ segmat, _Float16* __restrict__ Out) {
  constexpr int KCPI = 64, KPPI = 64, QPI = 72, PPI = 72;
  __shared__ __align__(16) _Float16 sKC[64 * KCPI];
  __shared__ __align__(16) _Float16 sVt[64 * KCPI];   // [s][r] (transposed)
  __shared__ __align__(16) _Float16 sKP[128 * KPPI];  // band of key_position
  __shared__ __align__(16) unsigned char uD[4 * 16 * 80 * 4];  // sQ / per-wave D
  __shared__ __align__(16) _Float16 sP[4 * 16 * PPI];
  __shared__ float sSeg[4][2][16];
  __shared__ __align__(16) _Float16 cbh[64];
  __shared__ __align__(16) _Float16 pbh[64];

  const int tid = threadIdx.x, lane = tid & 31, w = tid >> 5;
  const int b = blockIdx.z, h = blockIdx.y;
  const int qb = blockIdx.x * 64;
  const int q0 = qb + w * 16;
  const int mlc = lane & 15, hl = lane >> 4;

  _Float16* sQ = (_Float16*)uD;  // prologue only (overlaid with D region)

  if (tid < 64) {
    cbh[tid] = (_Float16)cb[h * cS + tid];
    pbh[tid] = (_Float16)pb[h * cS + tid];
  }
  {
    int idx = tid * 32;
    int r = idx >> 6, c = idx & 63;
    const _Float16* g = QC + ((size_t)(b * cQ + qb + r)) * cHID + h * cS + c;
    *(v8h*)&sQ[r * QPI + c + 0]  = *(const v8h*)(g + 0);
    *(v8h*)&sQ[r * QPI + c + 8]  = *(const v8h*)(g + 8);
    *(v8h*)&sQ[r * QPI + c + 16] = *(const v8h*)(g + 16);
    *(v8h*)&sQ[r * QPI + c + 24] = *(const v8h*)(g + 24);
  }
  __syncthreads();

  v16h aqc[2], aqp[2];
#pragma unroll
  for (int kk = 0; kk < 2; ++kk) {
    aqc[kk] = load_A_frag_bias(&sQ[(w * 16) * QPI], QPI, kk * 32, lane, cbh);
    aqp[kk] = load_A_frag_bias(&sQ[(w * 16) * QPI], QPI, kk * 32, lane, pbh);
  }
  {  // segment logits: lane -> (row = mlc, g = hl)
    float acc = 0.f;
    int qrow = w * 16 + mlc;
    for (int s = 0; s < cS; ++s)
      acc += ((float)sQ[qrow * QPI + s] + sb[h * cS + s]) *
             segenc[((size_t)hl * cH + h) * cS + s];
    sSeg[w][hl][mlc] = acc;
  }

  float mrow[8], lrow[8];
  v8f o[4] = {};
#pragma unroll
  for (int v = 0; v < 8; ++v) { mrow[v] = -1e30f; lrow[v] = 0.f; }

  float* sD = (float*)uD + w * (16 * 80);
  _Float16* sPw = &sP[w * 16 * PPI];

  for (int r0 = 0; r0 < cR; r0 += 64) {
    __syncthreads();
    {  // stage KC (natural) + V (transposed)
      int idx = tid * 32;
      int r = idx >> 6, c = idx & 63;
      const _Float16* gk = KC + ((size_t)(b * cR + r0 + r)) * cHID + h * cS + c;
      const _Float16* gv = Vp + ((size_t)(b * cR + r0 + r)) * cHID + h * cS + c;
      *(v8h*)&sKC[r * KCPI + c + 0]  = *(const v8h*)(gk + 0);
      *(v8h*)&sKC[r * KCPI + c + 8]  = *(const v8h*)(gk + 8);
      *(v8h*)&sKC[r * KCPI + c + 16] = *(const v8h*)(gk + 16);
      *(v8h*)&sKC[r * KCPI + c + 24] = *(const v8h*)(gk + 24);
#pragma unroll
      for (int j = 0; j < 32; ++j) sVt[(c + j) * KCPI + r] = gv[j];
    }
    {  // stage key_position band: 128 rows, one per thread
      int p = r0 + cQ - qb - 63 + tid;
      _Float16* dst = &sKP[tid * KPPI];
      if (p >= 0 && p < cPOS) {
        const _Float16* g = KPp + ((size_t)(b * cPOS + p)) * cHID + h * cS;
#pragma unroll
        for (int j = 0; j < 8; ++j) *(v8h*)&dst[j * 8] = *(const v8h*)&g[j * 8];
      } else {
        v8h z = {};
#pragma unroll
        for (int j = 0; j < 8; ++j) *(v8h*)&dst[j * 8] = z;
      }
    }
    __syncthreads();

    // content logits (4 col tiles x 2 k-steps)
    v8f ct[4];
#pragma unroll
    for (int t = 0; t < 4; ++t) {
      v8f a = {};
#pragma unroll
      for (int kk = 0; kk < 2; ++kk)
        a = wmma_f16(aqc[kk], load_B_frag(&sKC[(t * 16) * KCPI], KCPI, kk * 32, lane), a);
      ct[t] = a;
    }
    // position band D (5 col tiles of the 80-wide band), written to LDS
#pragma unroll
    for (int t = 0; t < 5; ++t) {
      int colb = 48 - 16 * w + t * 16;
      v8f a = {};
#pragma unroll
      for (int kk = 0; kk < 2; ++kk)
        a = wmma_f16(aqp[kk], load_B_frag(&sKP[colb * KPPI], KPPI, kk * 32, lane), a);
#pragma unroll
      for (int v = 0; v < 8; ++v)
        sD[(v + hl * 8) * 80 + t * 16 + mlc] = a[v];
    }
    // combine: content + shifted-pos + segment, scale, analytic causal mask
#pragma unroll
    for (int t = 0; t < 4; ++t) {
#pragma unroll
      for (int v = 0; v < 8; ++v) {
        int i = v + hl * 8, jj = t * 16 + mlc;
        int q = q0 + i, rr = r0 + jj;
        float lg = ct[t][v] + sD[i * 80 + (jj - i + 15)];
        lg += segmat[((size_t)b * cQ + q) * cR + rr] ? sSeg[w][1][i] : sSeg[w][0][i];
        lg *= kScale;
        if (rr > q + cM) lg += kNegInf;
        ct[t][v] = lg;
      }
    }
    // online softmax (row stats via 16-lane shuffle reductions)
#pragma unroll
    for (int v = 0; v < 8; ++v) {
      float tmax = fmaxf(fmaxf(ct[0][v], ct[1][v]), fmaxf(ct[2][v], ct[3][v]));
#pragma unroll
      for (int d = 1; d < 16; d <<= 1) tmax = fmaxf(tmax, __shfl_xor(tmax, d, 16));
      float mn = fmaxf(mrow[v], tmax);
      float corr = __expf(mrow[v] - mn);
      mrow[v] = mn;
      lrow[v] *= corr;
#pragma unroll
      for (int t = 0; t < 4; ++t) o[t][v] *= corr;
      float rs = 0.f;
#pragma unroll
      for (int t = 0; t < 4; ++t) { float e = __expf(ct[t][v] - mn); ct[t][v] = e; rs += e; }
#pragma unroll
      for (int d = 1; d < 16; d <<= 1) rs += __shfl_xor(rs, d, 16);
      lrow[v] += rs;
    }
    // P -> LDS (C layout -> A layout), then AV accumulation
#pragma unroll
    for (int t = 0; t < 4; ++t)
#pragma unroll
      for (int v = 0; v < 8; ++v)
        sPw[(v + hl * 8) * PPI + t * 16 + mlc] = (_Float16)ct[t][v];
#pragma unroll
    for (int kk = 0; kk < 2; ++kk) {
      v16h pa = load_A_frag(sPw, PPI, kk * 32, lane);
#pragma unroll
      for (int t = 0; t < 4; ++t)
        o[t] = wmma_f16(pa, load_B_frag(&sVt[(t * 16) * KCPI], KCPI, kk * 32, lane), o[t]);
    }
  }
#pragma unroll
  for (int t = 0; t < 4; ++t)
#pragma unroll
    for (int v = 0; v < 8; ++v) {
      int i = v + hl * 8, s = t * 16 + mlc;
      Out[((size_t)(b * cQ + q0 + i)) * cHID + h * cS + s] =
          (_Float16)(o[t][v] / lrow[v]);
    }
}

// ---------------------------------------------------------------------------
// Query-stream attention: only the P target rows (q = tpos[b,p]) matter.
// Pos logits gathered from precomputed PosQ.  grid: (P/64, H, B), block 128.
// ---------------------------------------------------------------------------
__global__ void __launch_bounds__(128) k_attn_query(
    const _Float16* __restrict__ QQ, const _Float16* __restrict__ KC,
    const _Float16* __restrict__ Vp, const _Float16* __restrict__ PosQ,
    const float* __restrict__ cb, const float* __restrict__ sb,
    const float* __restrict__ segenc, const unsigned char* __restrict__ segmat,
    const int* __restrict__ tpos, _Float16* __restrict__ Out) {
  constexpr int KCPI = 64, QPI = 72, PPI = 72;
  __shared__ __align__(16) _Float16 sQ[64 * QPI];
  __shared__ __align__(16) _Float16 sKC[64 * KCPI];
  __shared__ __align__(16) _Float16 sVt[64 * KCPI];
  __shared__ __align__(16) _Float16 sP[4 * 16 * PPI];
  __shared__ float sSeg[4][2][16];
  __shared__ int sTq[64];
  __shared__ __align__(16) _Float16 cbh[64];

  const int tid = threadIdx.x, lane = tid & 31, w = tid >> 5;
  const int b = blockIdx.z, h = blockIdx.y;
  const int pb0 = blockIdx.x * 64;
  const int p0 = pb0 + w * 16;
  const int mlc = lane & 15, hl = lane >> 4;

  if (tid < 64) {
    cbh[tid] = (_Float16)cb[h * cS + tid];
    sTq[tid] = tpos[b * cP + pb0 + tid];
  }
  {
    int idx = tid * 32;
    int r = idx >> 6, c = idx & 63;
    const _Float16* g = QQ + ((size_t)(b * cP + pb0 + r)) * cHID + h * cS + c;
    *(v8h*)&sQ[r * QPI + c + 0]  = *(const v8h*)(g + 0);
    *(v8h*)&sQ[r * QPI + c + 8]  = *(const v8h*)(g + 8);
    *(v8h*)&sQ[r * QPI + c + 16] = *(const v8h*)(g + 16);
    *(v8h*)&sQ[r * QPI + c + 24] = *(const v8h*)(g + 24);
  }
  __syncthreads();

  v16h aqc[2];
#pragma unroll
  for (int kk = 0; kk < 2; ++kk)
    aqc[kk] = load_A_frag_bias(&sQ[(w * 16) * QPI], QPI, kk * 32, lane, cbh);
  {
    float acc = 0.f;
    int qrow = w * 16 + mlc;
    for (int s = 0; s < cS; ++s)
      acc += ((float)sQ[qrow * QPI + s] + sb[h * cS + s]) *
             segenc[((size_t)hl * cH + h) * cS + s];
    sSeg[w][hl][mlc] = acc;
  }

  float mrow[8], lrow[8];
  v8f o[4] = {};
#pragma unroll
  for (int v = 0; v < 8; ++v) { mrow[v] = -1e30f; lrow[v] = 0.f; }
  _Float16* sPw = &sP[w * 16 * PPI];

  for (int r0 = 0; r0 < cR; r0 += 64) {
    __syncthreads();
    {
      int idx = tid * 32;
      int r = idx >> 6, c = idx & 63;
      const _Float16* gk = KC + ((size_t)(b * cR + r0 + r)) * cHID + h * cS + c;
      const _Float16* gv = Vp + ((size_t)(b * cR + r0 + r)) * cHID + h * cS + c;
      *(v8h*)&sKC[r * KCPI + c + 0]  = *(const v8h*)(gk + 0);
      *(v8h*)&sKC[r * KCPI + c + 8]  = *(const v8h*)(gk + 8);
      *(v8h*)&sKC[r * KCPI + c + 16] = *(const v8h*)(gk + 16);
      *(v8h*)&sKC[r * KCPI + c + 24] = *(const v8h*)(gk + 24);
#pragma unroll
      for (int j = 0; j < 32; ++j) sVt[(c + j) * KCPI + r] = gv[j];
    }
    __syncthreads();

    v8f ct[4];
#pragma unroll
    for (int t = 0; t < 4; ++t) {
      v8f a = {};
#pragma unroll
      for (int kk = 0; kk < 2; ++kk)
        a = wmma_f16(aqc[kk], load_B_frag(&sKC[(t * 16) * KCPI], KCPI, kk * 32, lane), a);
      ct[t] = a;
    }
#pragma unroll
    for (int t = 0; t < 4; ++t) {
#pragma unroll
      for (int v = 0; v < 8; ++v) {
        int i = v + hl * 8, jj = t * 16 + mlc;
        int prow = p0 + i, rr = r0 + jj;
        int tq = sTq[w * 16 + i];
        float lg = ct[t][v] +
                   (float)PosQ[(((size_t)(b * cH + h)) * cP + prow) * cPOS +
                               (rr - tq + cQ)];
        lg += segmat[((size_t)b * cQ + tq) * cR + rr] ? sSeg[w][1][i] : sSeg[w][0][i];
        lg *= kScale;
        if (rr >= tq + cM) lg += kNegInf;  // query mask includes diagonal
        ct[t][v] = lg;
      }
    }
#pragma unroll
    for (int v = 0; v < 8; ++v) {
      float tmax = fmaxf(fmaxf(ct[0][v], ct[1][v]), fmaxf(ct[2][v], ct[3][v]));
#pragma unroll
      for (int d = 1; d < 16; d <<= 1) tmax = fmaxf(tmax, __shfl_xor(tmax, d, 16));
      float mn = fmaxf(mrow[v], tmax);
      float corr = __expf(mrow[v] - mn);
      mrow[v] = mn;
      lrow[v] *= corr;
#pragma unroll
      for (int t = 0; t < 4; ++t) o[t][v] *= corr;
      float rs = 0.f;
#pragma unroll
      for (int t = 0; t < 4; ++t) { float e = __expf(ct[t][v] - mn); ct[t][v] = e; rs += e; }
#pragma unroll
      for (int d = 1; d < 16; d <<= 1) rs += __shfl_xor(rs, d, 16);
      lrow[v] += rs;
    }
#pragma unroll
    for (int t = 0; t < 4; ++t)
#pragma unroll
      for (int v = 0; v < 8; ++v)
        sPw[(v + hl * 8) * PPI + t * 16 + mlc] = (_Float16)ct[t][v];
#pragma unroll
    for (int kk = 0; kk < 2; ++kk) {
      v16h pa = load_A_frag(sPw, PPI, kk * 32, lane);
#pragma unroll
      for (int t = 0; t < 4; ++t)
        o[t] = wmma_f16(pa, load_B_frag(&sVt[(t * 16) * KCPI], KCPI, kk * 32, lane), o[t]);
    }
  }
#pragma unroll
  for (int t = 0; t < 4; ++t)
#pragma unroll
    for (int v = 0; v < 8; ++v) {
      int i = v + hl * 8, s = t * 16 + mlc;
      Out[((size_t)(b * cP + p0 + i)) * cHID + h * cS + s] =
          (_Float16)(o[t][v] / lrow[v]);
    }
}

// ---------------------------------------------------------------------------
// Host-side orchestration
// ---------------------------------------------------------------------------
extern "C" void kernel_launch(void* const* d_in, const int* in_sizes, int n_in,
                              void* d_out, int out_size, void* d_ws,
                              size_t ws_size, hipStream_t stream) {
  (void)in_sizes; (void)n_in; (void)out_size; (void)ws_size;

  const float* content_inputs    = (const float*)d_in[0];
  const float* query_inputs      = (const float*)d_in[1];
  const float* position_encoding = (const float*)d_in[2];
  const float* segment_encoding  = (const float*)d_in[3];
  const unsigned char* segment_matrix = (const unsigned char*)d_in[4];  // bool
  const float* target_mapping    = (const float*)d_in[5];
  // d_in[6]/d_in[7] (content_mask/query_mask) are causal: computed analytically.
  const float* content_bias  = (const float*)d_in[8];
  const float* position_bias = (const float*)d_in[9];
  const float* segment_bias  = (const float*)d_in[10];
  const float* mems          = (const float*)d_in[11];
  const float* Wq  = (const float*)d_in[12];
  const float* Wkc = (const float*)d_in[13];
  const float* Wv  = (const float*)d_in[14];
  const float* Wkp = (const float*)d_in[15];
  const float* Wo  = (const float*)d_in[16];

  uint8_t* wsp = (uint8_t*)d_ws;
  size_t off = 0;
  auto carve = [&](size_t bytes) -> void* {
    void* p = wsp + off;
    off += (bytes + 255) & ~size_t(255);
    return p;
  };
  _Float16* ctx16  = (_Float16*)carve((size_t)cB * cR * cHID * 2);
  _Float16* cont16 = (_Float16*)carve((size_t)cB * cQ * cHID * 2);
  _Float16* qry16  = (_Float16*)carve((size_t)cB * cP * cHID * 2);
  _Float16* pos16  = (_Float16*)carve((size_t)cB * cPOS * cHID * 2);
  _Float16* wq16   = (_Float16*)carve((size_t)cHID * cHID * 2);
  _Float16* wkc16  = (_Float16*)carve((size_t)cHID * cHID * 2);
  _Float16* wv16   = (_Float16*)carve((size_t)cHID * cHID * 2);
  _Float16* wkp16  = (_Float16*)carve((size_t)cHID * cHID * 2);
  _Float16* wo16   = (_Float16*)carve((size_t)cHID * cHID * 2);
  _Float16* KCp    = (_Float16*)carve((size_t)cB * cR * cHID * 2);
  _Float16* Vpp    = (_Float16*)carve((size_t)cB * cR * cHID * 2);
  _Float16* KPp    = (_Float16*)carve((size_t)cB * cPOS * cHID * 2);
  _Float16* QCp    = (_Float16*)carve((size_t)cB * cQ * cHID * 2);
  _Float16* QQp    = (_Float16*)carve((size_t)cB * cP * cHID * 2);
  _Float16* PosQ   = (_Float16*)carve((size_t)cB * cH * cP * cPOS * 2);
  _Float16* AC     = (_Float16*)carve((size_t)cB * cQ * cHID * 2);
  _Float16* AQ     = (_Float16*)carve((size_t)cB * cP * cHID * 2);
  int* tpos        = (int*)carve((size_t)cB * cP * 4);

  const dim3 blk(128);

  // ---- f32 -> f16 conversions ----
  k_cvt<<<dim3(4096), dim3(256), 0, stream>>>(position_encoding, pos16,
                                              (size_t)cB * cPOS * cHID);
  k_cvt<<<dim3(512), dim3(256), 0, stream>>>(query_inputs, qry16,
                                             (size_t)cB * cP * cHID);
  k_cvt<<<dim3(2048), dim3(256), 0, stream>>>(content_inputs, cont16,
                                              (size_t)cB * cQ * cHID);
  k_cvt<<<dim3(1024), dim3(256), 0, stream>>>(Wq, wq16, (size_t)cHID * cHID);
  k_cvt<<<dim3(1024), dim3(256), 0, stream>>>(Wkc, wkc16, (size_t)cHID * cHID);
  k_cvt<<<dim3(1024), dim3(256), 0, stream>>>(Wv, wv16, (size_t)cHID * cHID);
  k_cvt<<<dim3(1024), dim3(256), 0, stream>>>(Wkp, wkp16, (size_t)cHID * cHID);
  k_cvt<<<dim3(1024), dim3(256), 0, stream>>>(Wo, wo16, (size_t)cHID * cHID);
  k_concat<<<dim3(4096), dim3(256), 0, stream>>>(mems, content_inputs, ctx16,
                                                 (size_t)cB * cR * cHID);
  k_tpos<<<dim3(2), dim3(256), 0, stream>>>(target_mapping, tpos);

  // ---- projections (WMMA GEMMs) ----
  k_gemm<0><<<dim3(cHID / 64, (cB * cR) / 64), blk, 0, stream>>>(
      ctx16, wkc16, KCp, cB * cR, cHID, cHID);
  k_gemm<0><<<dim3(cHID / 64, (cB * cR) / 64), blk, 0, stream>>>(
      ctx16, wv16, Vpp, cB * cR, cHID, cHID);
  k_gemm<0><<<dim3(cHID / 64, (cB * cPOS) / 64), blk, 0, stream>>>(
      pos16, wkp16, KPp, cB * cPOS, cHID, cHID);
  k_gemm<0><<<dim3(cHID / 64, (cB * cQ) / 64), blk, 0, stream>>>(
      cont16, wq16, QCp, cB * cQ, cHID, cHID);
  k_gemm<0><<<dim3(cHID / 64, (cB * cP) / 64), blk, 0, stream>>>(
      qry16, wq16, QQp, cB * cP, cHID, cHID);

  // ---- query-stream positional logits (full Rp GEMM, gathered later) ----
  k_posq<<<dim3(cPOS / 64, cP / 64, cB * cH), blk, 0, stream>>>(
      QQp, KPp, position_bias, PosQ);

  // ---- attention ----
  k_attn_content<<<dim3(cQ / 64, cH, cB), blk, 0, stream>>>(
      QCp, KCp, Vpp, KPp, content_bias, position_bias, segment_bias,
      segment_encoding, segment_matrix, AC);
  k_attn_query<<<dim3(cP / 64, cH, cB), blk, 0, stream>>>(
      QQp, KCp, Vpp, PosQ, content_bias, segment_bias, segment_encoding,
      segment_matrix, tpos, AQ);

  // ---- output merges (f32 out) ----
  k_gemm<1><<<dim3(cHID / 64, (cB * cQ) / 64), blk, 0, stream>>>(
      AC, wo16, d_out, cB * cQ, cHID, cHID);
  k_gemm<1><<<dim3(cHID / 64, (cB * cP) / 64), blk, 0, stream>>>(
      AQ, wo16, (float*)d_out + (size_t)cB * cQ * cHID, cB * cP, cHID, cHID);
}
